// TAM_52518860095742
// MI455X (gfx1250) — matrix operand order repaired and checked by reading, not compiled, not run on hardware
//
#include <hip/hip_runtime.h>
#include <hip/hip_bf16.h>
#include <math.h>

// ---------------- types for WMMA bf16 ----------------
typedef unsigned short u16b;
typedef __attribute__((ext_vector_type(16))) __bf16 v16bf;
typedef __attribute__((ext_vector_type(8)))  float  v8f;

union FragBF { uint4 u[2]; v16bf v; };
static_assert(sizeof(FragBF) == 32, "frag size");

__device__ __forceinline__ u16b f2bf(float f) {
  unsigned u = __float_as_uint(f);
  u += 0x7fffu + ((u >> 16) & 1u);           // round-to-nearest-even
  return (u16b)(u >> 16);
}
__device__ __forceinline__ float sigf(float x) { return 1.f / (1.f + __expf(-x)); }

// ---------------- problem constants ----------------
#define S_N   600
#define S_PAD 640
#define L_N   40
#define P_N   100
#define P_PAD 128
#define I_N   512
#define HS_N  128     // scene hidden
#define HP_N  256     // plot hidden
#define HSC_N 256     // scene-level hidden
#define U_N   1536
#define MH    3072
#define W1K   6145
#define NTP   5
#define WWIN  60

// =====================================================
// Generic bf16 WMMA core: block=128thr (4 waves, 2x2), 64x64 tile
// A: (M x K) bf16 row-major, W: (N x K) bf16 row-major (computes A @ W^T)
// =====================================================
__device__ __forceinline__ void gemm_core(const u16b* __restrict__ A, int lda,
                                          const u16b* __restrict__ W, int ldw,
                                          int K, int m0, int n0, v8f acc[2][2]) {
  const int lane = threadIdx.x & 31;
  const int half = lane >> 4;
  const int r16  = lane & 15;
  const u16b* arow0 = A + (size_t)(m0 + r16) * lda;
  const u16b* arow1 = arow0 + (size_t)16 * lda;
  const u16b* wrow0 = W + (size_t)(n0 + r16) * ldw;
  const u16b* wrow1 = wrow0 + (size_t)16 * ldw;
  for (int k0 = 0; k0 < K; k0 += 32) {
    FragBF a0, a1, b0, b1;
    const u16b* pa0 = arow0 + k0 + half * 8;
    const u16b* pa1 = arow1 + k0 + half * 8;
    const u16b* pb0 = wrow0 + k0 + half * 8;
    const u16b* pb1 = wrow1 + k0 + half * 8;
    a0.u[0] = *(const uint4*)pa0;       a0.u[1] = *(const uint4*)(pa0 + 16);
    a1.u[0] = *(const uint4*)pa1;       a1.u[1] = *(const uint4*)(pa1 + 16);
    b0.u[0] = *(const uint4*)pb0;       b0.u[1] = *(const uint4*)(pb0 + 16);
    b1.u[0] = *(const uint4*)pb1;       b1.u[1] = *(const uint4*)(pb1 + 16);
    __builtin_prefetch(pa0 + 64, 0, 1);
    __builtin_prefetch(pb0 + 64, 0, 1);
    acc[0][0] = __builtin_amdgcn_wmma_f32_16x16x32_bf16(false, a0.v, false, b0.v, (short)0, acc[0][0], false, false);
    acc[0][1] = __builtin_amdgcn_wmma_f32_16x16x32_bf16(false, a0.v, false, b1.v, (short)0, acc[0][1], false, false);
    acc[1][0] = __builtin_amdgcn_wmma_f32_16x16x32_bf16(false, a1.v, false, b0.v, (short)0, acc[1][0], false, false);
    acc[1][1] = __builtin_amdgcn_wmma_f32_16x16x32_bf16(false, a1.v, false, b1.v, (short)0, acc[1][1], false, false);
  }
}

__device__ __forceinline__ void acc_zero(v8f acc[2][2]) {
  v8f z = {0.f,0.f,0.f,0.f,0.f,0.f,0.f,0.f};
  acc[0][0] = z; acc[0][1] = z; acc[1][0] = z; acc[1][1] = z;
}

// C = A@W^T (+ bias[n]) (+ ADD[m*ldadd+n]); rows >= Mreal not stored
__global__ void k_gemm_store(float* __restrict__ C, int ldc,
                             const u16b* __restrict__ A, int lda,
                             const u16b* __restrict__ W, int ldw,
                             const float* __restrict__ bias,
                             const float* __restrict__ ADD, int ldadd,
                             int Mreal, int K) {
  const int wave = threadIdx.x >> 5, wm = wave >> 1, wn = wave & 1;
  const int m0 = blockIdx.y * 64 + wm * 32;
  const int n0 = blockIdx.x * 64 + wn * 32;
  v8f acc[2][2]; acc_zero(acc);
  gemm_core(A, lda, W, ldw, K, m0, n0, acc);
  const int lane = threadIdx.x & 31, half = lane >> 4, r16 = lane & 15;
#pragma unroll
  for (int i = 0; i < 2; ++i)
#pragma unroll
    for (int j = 0; j < 2; ++j) {
      const int gn = n0 + j * 16 + r16;
#pragma unroll
      for (int r = 0; r < 8; ++r) {
        const int gm = m0 + i * 16 + half * 8 + r;
        if (gm < Mreal) {
          float v = acc[i][j][r];
          if (bias) v += bias[gn];
          if (ADD)  v += ADD[(size_t)gm * ldadd + gn];
          C[(size_t)gm * ldc + gn] = v;
        }
      }
    }
}

// attention: scores[m] += sum_n tanh((A@W^T)[m,n] + b[n]) * v[n]
__global__ void k_gemm_att(float* __restrict__ scores,
                           const u16b* __restrict__ A, const u16b* __restrict__ W,
                           const float* __restrict__ attb, const float* __restrict__ attv) {
  __shared__ float part[64];
  const int wave = threadIdx.x >> 5, wm = wave >> 1, wn = wave & 1;
  const int m0 = blockIdx.y * 64 + wm * 32;
  const int n0 = blockIdx.x * 64 + wn * 32;
  v8f acc[2][2]; acc_zero(acc);
  gemm_core(A, 256, W, 256, 256, m0, n0, acc);
  if (threadIdx.x < 64) part[threadIdx.x] = 0.f;
  __syncthreads();
  const int lane = threadIdx.x & 31, half = lane >> 4, r16 = lane & 15;
#pragma unroll
  for (int i = 0; i < 2; ++i)
#pragma unroll
    for (int j = 0; j < 2; ++j) {
      const int gn = n0 + j * 16 + r16;
#pragma unroll
      for (int r = 0; r < 8; ++r) {
        const int lm = wm * 32 + i * 16 + half * 8 + r;
        float t = tanhf(acc[i][j][r] + attb[gn]);
        atomicAdd(&part[lm], t * attv[gn]);
      }
    }
  __syncthreads();
  if (threadIdx.x < 64)
    atomicAdd(&scores[(size_t)blockIdx.y * 64 + threadIdx.x], part[threadIdx.x]);
}

// fused MLP product-GEMM + ReLU + dot(Wo) -> y[p,s]
__global__ void k_gemm_y(float* __restrict__ y,
                         const u16b* __restrict__ pu, const u16b* __restrict__ Wpr,
                         const float* __restrict__ H1, const float* __restrict__ basep,
                         const float* __restrict__ cosv, const float* __restrict__ w1c,
                         const float* __restrict__ Wo) {
  __shared__ float part[64];
  const int p = blockIdx.z;
  const u16b* A = pu + (size_t)p * S_PAD * U_N;
  const int wave = threadIdx.x >> 5, wm = wave >> 1, wn = wave & 1;
  const int m0 = blockIdx.y * 64 + wm * 32;
  const int n0 = blockIdx.x * 64 + wn * 32;
  v8f acc[2][2]; acc_zero(acc);
  gemm_core(A, U_N, Wpr, U_N, U_N, m0, n0, acc);
  if (threadIdx.x < 64) part[threadIdx.x] = 0.f;
  __syncthreads();
  const int lane = threadIdx.x & 31, half = lane >> 4, r16 = lane & 15;
#pragma unroll
  for (int i = 0; i < 2; ++i)
#pragma unroll
    for (int j = 0; j < 2; ++j) {
      const int gn = n0 + j * 16 + r16;
#pragma unroll
      for (int r = 0; r < 8; ++r) {
        const int gm = m0 + i * 16 + half * 8 + r;
        if (gm < S_N) {
          float v = acc[i][j][r] + H1[(size_t)gm * MH + gn] + basep[p * MH + gn]
                  + cosv[p * S_N + gm] * w1c[gn];
          v = fmaxf(v, 0.f) * Wo[gn];
          atomicAdd(&part[wm * 32 + i * 16 + half * 8 + r], v);
        }
      }
    }
  __syncthreads();
  const int s = blockIdx.y * 64 + threadIdx.x;
  if (threadIdx.x < 64 && s < S_N) atomicAdd(&y[p * S_N + s], part[threadIdx.x]);
}

// =====================================================
// pointwise / sequential kernels
// =====================================================
__global__ void k_cvt(u16b* __restrict__ dst, const float* __restrict__ src, long n) {
  long i = (long)blockIdx.x * 256 + threadIdx.x;
  if (i < n) dst[i] = f2bf(src[i]);
}

__global__ void k_lstm_point(const float* __restrict__ zh, float* __restrict__ c,
                             u16b* __restrict__ hbf, float* __restrict__ so,
                             int tout, int dofs) {
  int idx = blockIdx.x * 256 + threadIdx.x;
  if (idx >= S_N * HS_N) return;
  int s = idx >> 7, d = idx & 127;
  const float* z = zh + (size_t)s * 512;
  float gi = sigf(z[d]), gf = sigf(z[128 + d]);
  float gg = tanhf(z[256 + d]), go = sigf(z[384 + d]);
  float cv = gf * c[idx] + gi * gg; c[idx] = cv;
  float h = go * tanhf(cv);
  hbf[(size_t)s * HS_N + d] = f2bf(h);
  so[((size_t)s * L_N + tout) * 256 + dofs + d] = h;
}

// batch-1 BiLSTM (H=256, gates=1024), whole sequence in one kernel, grid=2 (dir)
__global__ void k_seq_lstm(const float* __restrict__ xw_f, const float* __restrict__ xw_b,
                           const float* __restrict__ U_f, const float* __restrict__ U_b,
                           float* __restrict__ out, int T) {
  const int dir = blockIdx.x;
  const float* xw = dir ? xw_b : xw_f;
  const float* U  = dir ? U_b  : U_f;
  __shared__ float hsh[256];
  __shared__ float csh[256];
  __shared__ float zsh[1024];
  const int g = threadIdx.x;             // 1024 threads
  if (g < 256) { hsh[g] = 0.f; csh[g] = 0.f; }
  __syncthreads();
  const float* u = U + (size_t)g * 256;
  for (int k = 0; k < T; ++k) {
    const int row = dir ? (T - 1 - k) : k;
    float a = xw[(size_t)row * 1024 + g];
    for (int j = 0; j < 256; ++j) a += hsh[j] * u[j];
    zsh[g] = a;
    __syncthreads();
    if (g < 256) {
      float gi = sigf(zsh[g]), gf = sigf(zsh[256 + g]);
      float gg = tanhf(zsh[512 + g]), go = sigf(zsh[768 + g]);
      float cv = gf * csh[g] + gi * gg; csh[g] = cv;
      float h = go * tanhf(cv); hsh[g] = h;
      out[(size_t)row * 512 + dir * 256 + g] = h;
    }
    __syncthreads();
  }
}

__global__ void k_softmax_scene(const float* __restrict__ scores, const float* __restrict__ so,
                                const int* __restrict__ lens, float* __restrict__ semb) {
  __shared__ float sc[L_N];
  const int s = blockIdx.x, tid = threadIdx.x;
  if (tid < L_N) sc[tid] = scores[s * L_N + tid];
  __syncthreads();
  if (tid == 0) {
    int len = lens[s]; len = len < 1 ? 1 : (len > L_N ? L_N : len);
    float mx = -3.4e38f;
    for (int l = 0; l < len; ++l) mx = fmaxf(mx, sc[l]);
    float sum = 0.f;
    for (int l = 0; l < len; ++l) { float e = __expf(sc[l] - mx); sc[l] = e; sum += e; }
    float inv = 1.f / sum;
    for (int l = 0; l < len; ++l) sc[l] *= inv;
    for (int l = len; l < L_N; ++l) sc[l] = 0.f;
  }
  __syncthreads();
  float a = 0.f;
  const float* base = so + (size_t)s * L_N * 256 + tid;
  for (int l = 0; l < L_N; ++l) a += sc[l] * base[l * 256];
  semb[(size_t)s * 256 + tid] = a;
}

__global__ void k_cumsum(const float* __restrict__ emb, float* __restrict__ cs) {
  const int d = threadIdx.x;           // 512 threads, 1 block
  float run = 0.f;
  cs[d] = 0.f;
  for (int s = 0; s < S_N; ++s) { run += emb[(size_t)s * 512 + d]; cs[(size_t)(s + 1) * 512 + d] = run; }
}

__global__ void k_build_u(const float* __restrict__ emb, const float* __restrict__ cs,
                          float* __restrict__ u) {
  int idx = blockIdx.x * 256 + threadIdx.x;
  if (idx >= S_N * 512) return;
  int s = idx >> 9, d = idx & 511;
  float c_s = cs[(size_t)s * 512 + d];
  int lo = s - WWIN; if (lo < 0) lo = 0;
  float lsum = c_s - cs[(size_t)lo * 512 + d];
  int lcnt = s < WWIN ? s : WWIN; if (lcnt < 1) lcnt = 1;
  float left = (s == 0) ? 0.f : lsum / (float)lcnt;
  int re = s + WWIN; if (re > S_N) re = S_N;
  float rsum = cs[(size_t)re * 512 + d] - c_s;
  float right = (s == S_N - 1) ? 0.f : rsum / (float)(re - s);
  float* ur = u + (size_t)s * U_N;
  ur[d] = emb[idx]; ur[512 + d] = left; ur[1024 + d] = right;
}

__global__ void k_gather_tp(const float* __restrict__ pe, const int* __restrict__ ids,
                            float* __restrict__ tpe) {
  int idx = blockIdx.x * 256 + threadIdx.x;
  if (idx >= NTP * 512) return;
  int p = idx >> 9, d = idx & 511;
  tpe[idx] = pe[(size_t)ids[p] * 512 + d];
}

__global__ void k_tpproj(const float* __restrict__ tpe, const float* __restrict__ Wp,
                         const float* __restrict__ bp, float* __restrict__ tpp) {
  int idx = blockIdx.x * 128 + threadIdx.x;
  if (idx >= NTP * U_N) return;
  int p = idx / U_N, j = idx % U_N;
  const float* w = Wp + (size_t)j * 512;
  const float* x = tpe + (size_t)p * 512;
  float a = bp[j];
  for (int k = 0; k < 512; ++k) a += x[k] * w[k];
  tpp[idx] = a;
}

__global__ void k_basep(const float* __restrict__ tpp, const float* __restrict__ W1,
                        const float* __restrict__ b1, float* __restrict__ basep) {
  int idx = blockIdx.x * 256 + threadIdx.x;
  if (idx >= NTP * MH) return;
  int p = idx / MH, j = idx % MH;
  const float* row = W1 + (size_t)j * W1K;
  const float* x = tpp + (size_t)p * U_N;
  float a = b1[j];
  for (int k = 0; k < U_N; ++k) a += x[k] * (row[1536 + k] + row[4608 + k]);
  basep[idx] = a;
}

__global__ void k_w1c(const float* __restrict__ W1, float* __restrict__ w1c) {
  int j = blockIdx.x * 256 + threadIdx.x;
  if (j < MH) w1c[j] = W1[(size_t)j * W1K + 6144];
}

__global__ void k_prepW(u16b* __restrict__ Wu, u16b* __restrict__ Wpr,
                        const float* __restrict__ W1) {
  long i = (long)blockIdx.x * 256 + threadIdx.x;
  if (i >= (long)MH * U_N) return;
  long j = i / U_N, k = i % U_N;
  const float* row = W1 + j * W1K;
  Wu[i]  = f2bf(row[k] - row[4608 + k]);
  Wpr[i] = f2bf(row[3072 + k]);
}

__global__ void k_rownorm(float* __restrict__ out, const float* __restrict__ src, int len) {
  __shared__ float sh[256];
  const float* r = src + (size_t)blockIdx.x * len;
  float a = 0.f;
  for (int i = threadIdx.x; i < len; i += 256) { float v = r[i]; a += v * v; }
  sh[threadIdx.x] = a; __syncthreads();
  for (int s = 128; s > 0; s >>= 1) { if (threadIdx.x < s) sh[threadIdx.x] += sh[threadIdx.x + s]; __syncthreads(); }
  if (threadIdx.x == 0) out[blockIdx.x] = fmaxf(sqrtf(sh[0]), 1e-8f);
}

__global__ void k_cos(const float* __restrict__ tpp, const float* __restrict__ u,
                      const float* __restrict__ pn, const float* __restrict__ sn,
                      float* __restrict__ cosv) {
  __shared__ float sh[128];
  int p = blockIdx.x / S_N, s = blockIdx.x % S_N;
  const float* a = tpp + (size_t)p * U_N;
  const float* b = u + (size_t)s * U_N;
  float acc = 0.f;
  for (int i = threadIdx.x; i < U_N; i += 128) acc += a[i] * b[i];
  sh[threadIdx.x] = acc; __syncthreads();
  for (int t = 64; t > 0; t >>= 1) { if (threadIdx.x < t) sh[threadIdx.x] += sh[threadIdx.x + t]; __syncthreads(); }
  if (threadIdx.x == 0) cosv[blockIdx.x] = sh[0] / (pn[p] * sn[s]);
}

__global__ void k_build_pu(const float* __restrict__ tpp, const float* __restrict__ u,
                           u16b* __restrict__ pu) {
  long idx = (long)blockIdx.x * 256 + threadIdx.x;
  if (idx >= (long)NTP * S_PAD * U_N) return;
  int k = idx % U_N; long rem = idx / U_N;
  int s = rem % S_PAD, p = rem / S_PAD;
  pu[idx] = f2bf(tpp[(size_t)p * U_N + k] * u[(size_t)s * U_N + k]);
}

__global__ void k_inity(float* __restrict__ y, const float* __restrict__ bo) {
  int i = blockIdx.x * 256 + threadIdx.x;
  if (i < NTP * S_N) y[i] = bo[0];
}

// =====================================================
// host orchestration
// =====================================================
extern "C" void kernel_launch(void* const* d_in, const int* in_sizes, int n_in,
                              void* d_out, int out_size, void* d_ws, size_t ws_size,
                              hipStream_t stream) {
  (void)in_sizes; (void)n_in; (void)out_size; (void)ws_size;
  const float* plot     = (const float*)d_in[0];
  const float* script   = (const float*)d_in[1];
  const int*   tp_ids   = (const int*)d_in[2];
  const int*   lens     = (const int*)d_in[3];
  const float* peWihf   = (const float*)d_in[4];
  const float* peWhhf   = (const float*)d_in[5];
  const float* pebf     = (const float*)d_in[6];
  const float* peWihb   = (const float*)d_in[7];
  const float* peWhhb   = (const float*)d_in[8];
  const float* pebb     = (const float*)d_in[9];
  const float* seWihf   = (const float*)d_in[10];
  const float* seWhhf   = (const float*)d_in[11];
  const float* sebf     = (const float*)d_in[12];
  const float* seWihb   = (const float*)d_in[13];
  const float* seWhhb   = (const float*)d_in[14];
  const float* sebb     = (const float*)d_in[15];
  const float* scWihf   = (const float*)d_in[16];
  const float* scWhhf   = (const float*)d_in[17];
  const float* scbf     = (const float*)d_in[18];
  const float* scWihb   = (const float*)d_in[19];
  const float* scWhhb   = (const float*)d_in[20];
  const float* scbb     = (const float*)d_in[21];
  const float* attW     = (const float*)d_in[22];
  const float* attb     = (const float*)d_in[23];
  const float* attv     = (const float*)d_in[24];
  const float* projW    = (const float*)d_in[25];
  const float* projb    = (const float*)d_in[26];
  const float* mlpW1    = (const float*)d_in[27];
  const float* mlpb1    = (const float*)d_in[28];
  const float* mlpWo    = (const float*)d_in[29];
  const float* mlpbo    = (const float*)d_in[30];
  float* y = (float*)d_out;

  // ---- workspace carve-up ----
  size_t off = 0;
  auto alloc = [&](size_t bytes) -> char* {
    char* p = (char*)d_ws + off;
    off += (bytes + 255) & ~(size_t)255;
    return p;
  };
  u16b* script_bf = (u16b*)alloc((size_t)24000 * 512 * 2);
  u16b* seWihf_bf = (u16b*)alloc((size_t)512 * 512 * 2);
  u16b* seWihb_bf = (u16b*)alloc((size_t)512 * 512 * 2);
  u16b* seWhhf_bf = (u16b*)alloc((size_t)512 * 128 * 2);
  u16b* seWhhb_bf = (u16b*)alloc((size_t)512 * 128 * 2);
  float* xw_f     = (float*)alloc((size_t)24000 * 512 * 4);
  float* xw_b     = (float*)alloc((size_t)24000 * 512 * 4);
  u16b* hbf_f     = (u16b*)alloc((size_t)S_PAD * 128 * 2);
  u16b* hbf_b     = (u16b*)alloc((size_t)S_PAD * 128 * 2);
  float* c_f      = (float*)alloc((size_t)S_N * 128 * 4);
  float* c_b      = (float*)alloc((size_t)S_N * 128 * 4);
  float* zh_f     = (float*)alloc((size_t)S_PAD * 512 * 4);
  float* zh_b     = (float*)alloc((size_t)S_PAD * 512 * 4);
  float* so       = (float*)alloc((size_t)24000 * 256 * 4);
  u16b* so_bf     = (u16b*)alloc((size_t)24000 * 256 * 2);
  u16b* attW_bf   = (u16b*)alloc((size_t)256 * 256 * 2);
  float* scores   = (float*)alloc((size_t)24000 * 4);
  float* semb     = (float*)alloc((size_t)S_PAD * 256 * 4);
  u16b* semb_bf   = (u16b*)alloc((size_t)S_PAD * 256 * 2);
  u16b* scWihf_bf = (u16b*)alloc((size_t)1024 * 256 * 2);
  u16b* scWihb_bf = (u16b*)alloc((size_t)1024 * 256 * 2);
  float* xwsc_f   = (float*)alloc((size_t)S_PAD * 1024 * 4);
  float* xwsc_b   = (float*)alloc((size_t)S_PAD * 1024 * 4);
  float* emb      = (float*)alloc((size_t)S_N * 512 * 4);
  float* cs       = (float*)alloc((size_t)(S_N + 1) * 512 * 4);
  float* u        = (float*)alloc((size_t)S_PAD * U_N * 4);
  u16b* u_bf      = (u16b*)alloc((size_t)S_PAD * U_N * 2);
  u16b* plot_bf   = (u16b*)alloc((size_t)P_PAD * 512 * 2);
  u16b* peWihf_bf = (u16b*)alloc((size_t)1024 * 512 * 2);
  u16b* peWihb_bf = (u16b*)alloc((size_t)1024 * 512 * 2);
  float* xwp_f    = (float*)alloc((size_t)P_PAD * 1024 * 4);
  float* xwp_b    = (float*)alloc((size_t)P_PAD * 1024 * 4);
  float* pemb     = (float*)alloc((size_t)P_N * 512 * 4);
  float* tpe      = (float*)alloc((size_t)NTP * 512 * 4);
  float* tpp      = (float*)alloc((size_t)NTP * U_N * 4);
  u16b* Wu_bf     = (u16b*)alloc((size_t)MH * U_N * 2);
  u16b* Wpr_bf    = (u16b*)alloc((size_t)MH * U_N * 2);
  float* H1       = (float*)alloc((size_t)S_PAD * MH * 4);
  float* basep    = (float*)alloc((size_t)NTP * MH * 4);
  float* w1c      = (float*)alloc((size_t)MH * 4);
  u16b* pu        = (u16b*)alloc((size_t)NTP * S_PAD * U_N * 2);
  float* pn       = (float*)alloc((size_t)NTP * 4);
  float* sn       = (float*)alloc((size_t)S_N * 4);
  float* cosv     = (float*)alloc((size_t)NTP * S_N * 4);

  auto GR = [](long n) { return dim3((unsigned)((n + 255) / 256)); };

  // ---- zero-init (pads + accumulators) ----
  hipMemsetAsync(hbf_f, 0, (size_t)S_PAD * 128 * 2, stream);
  hipMemsetAsync(hbf_b, 0, (size_t)S_PAD * 128 * 2, stream);
  hipMemsetAsync(c_f, 0, (size_t)S_N * 128 * 4, stream);
  hipMemsetAsync(c_b, 0, (size_t)S_N * 128 * 4, stream);
  hipMemsetAsync(scores, 0, (size_t)24000 * 4, stream);
  hipMemsetAsync(semb, 0, (size_t)S_PAD * 256 * 4, stream);
  hipMemsetAsync(u, 0, (size_t)S_PAD * U_N * 4, stream);
  hipMemsetAsync(plot_bf, 0, (size_t)P_PAD * 512 * 2, stream);

  // ---- bf16 conversions ----
  k_cvt<<<GR(24000L * 512), 256, 0, stream>>>(script_bf, script, 24000L * 512);
  k_cvt<<<GR(512L * 512), 256, 0, stream>>>(seWihf_bf, seWihf, 512L * 512);
  k_cvt<<<GR(512L * 512), 256, 0, stream>>>(seWihb_bf, seWihb, 512L * 512);
  k_cvt<<<GR(512L * 128), 256, 0, stream>>>(seWhhf_bf, seWhhf, 512L * 128);
  k_cvt<<<GR(512L * 128), 256, 0, stream>>>(seWhhb_bf, seWhhb, 512L * 128);
  k_cvt<<<GR(256L * 256), 256, 0, stream>>>(attW_bf, attW, 256L * 256);
  k_cvt<<<GR(1024L * 256), 256, 0, stream>>>(scWihf_bf, scWihf, 1024L * 256);
  k_cvt<<<GR(1024L * 256), 256, 0, stream>>>(scWihb_bf, scWihb, 1024L * 256);
  k_cvt<<<GR(1024L * 512), 256, 0, stream>>>(peWihf_bf, peWihf, 1024L * 512);
  k_cvt<<<GR(1024L * 512), 256, 0, stream>>>(peWihb_bf, peWihb, 1024L * 512);
  k_cvt<<<GR((long)P_N * 512), 256, 0, stream>>>(plot_bf, plot, (long)P_N * 512);

  // ---- hoisted input projections (WMMA) ----
  k_gemm_store<<<dim3(8, 375), 128, 0, stream>>>(xw_f, 512, script_bf, 512, seWihf_bf, 512, sebf, nullptr, 0, 24000, 512);
  k_gemm_store<<<dim3(8, 375), 128, 0, stream>>>(xw_b, 512, script_bf, 512, seWihb_bf, 512, sebb, nullptr, 0, 24000, 512);
  k_gemm_store<<<dim3(16, 2), 128, 0, stream>>>(xwp_f, 1024, plot_bf, 512, peWihf_bf, 512, pebf, nullptr, 0, P_PAD, 512);
  k_gemm_store<<<dim3(16, 2), 128, 0, stream>>>(xwp_b, 1024, plot_bf, 512, peWihb_bf, 512, pebb, nullptr, 0, P_PAD, 512);

  // ---- plot BiLSTM (batch 1, sequential) ----
  k_seq_lstm<<<2, 1024, 0, stream>>>(xwp_f, xwp_b, peWhhf, peWhhb, pemb, P_N);

  // ---- scene BiLSTM recurrence: 40 steps, WMMA per step ----
  for (int t = 0; t < L_N; ++t) {
    k_gemm_store<<<dim3(8, 10), 128, 0, stream>>>(zh_f, 512, hbf_f, 128, seWhhf_bf, 128,
                                                  nullptr, xw_f + (size_t)t * 512, L_N * 512, S_N, 128);
    k_gemm_store<<<dim3(8, 10), 128, 0, stream>>>(zh_b, 512, hbf_b, 128, seWhhb_bf, 128,
                                                  nullptr, xw_b + (size_t)(L_N - 1 - t) * 512, L_N * 512, S_N, 128);
    k_lstm_point<<<GR((long)S_N * 128), 256, 0, stream>>>(zh_f, c_f, hbf_f, so, t, 0);
    k_lstm_point<<<GR((long)S_N * 128), 256, 0, stream>>>(zh_b, c_b, hbf_b, so, L_N - 1 - t, 128);
  }

  // ---- attention ----
  k_cvt<<<GR(24000L * 256), 256, 0, stream>>>(so_bf, so, 24000L * 256);
  k_gemm_att<<<dim3(4, 375), 128, 0, stream>>>(scores, so_bf, attW_bf, attb, attv);
  k_softmax_scene<<<S_N, 256, 0, stream>>>(scores, so, lens, semb);

  // ---- scene-level BiLSTM ----
  k_cvt<<<GR((long)S_PAD * 256), 256, 0, stream>>>(semb_bf, semb, (long)S_PAD * 256);
  k_gemm_store<<<dim3(16, 10), 128, 0, stream>>>(xwsc_f, 1024, semb_bf, 256, scWihf_bf, 256, scbf, nullptr, 0, S_PAD, 256);
  k_gemm_store<<<dim3(16, 10), 128, 0, stream>>>(xwsc_b, 1024, semb_bf, 256, scWihb_bf, 256, scbb, nullptr, 0, S_PAD, 256);
  k_seq_lstm<<<2, 1024, 0, stream>>>(xwsc_f, xwsc_b, scWhhf, scWhhb, emb, S_N);

  // ---- context windows + u ----
  k_cumsum<<<1, 512, 0, stream>>>(emb, cs);
  k_build_u<<<GR((long)S_N * 512), 256, 0, stream>>>(emb, cs, u);
  k_cvt<<<GR((long)S_PAD * U_N), 256, 0, stream>>>(u_bf, u, (long)S_PAD * U_N);

  // ---- turning-point path ----
  k_gather_tp<<<GR((long)NTP * 512), 256, 0, stream>>>(pemb, tp_ids, tpe);
  k_tpproj<<<dim3((NTP * U_N + 127) / 128), 128, 0, stream>>>(tpe, projW, projb, tpp);

  // ---- MLP (algebraic feat decomposition) ----
  k_prepW<<<GR((long)MH * U_N), 256, 0, stream>>>(Wu_bf, Wpr_bf, mlpW1);
  k_gemm_store<<<dim3(48, 10), 128, 0, stream>>>(H1, MH, u_bf, U_N, Wu_bf, U_N, nullptr, nullptr, 0, S_PAD, U_N);
  k_basep<<<GR((long)NTP * MH), 256, 0, stream>>>(tpp, mlpW1, mlpb1, basep);
  k_w1c<<<GR(MH), 256, 0, stream>>>(mlpW1, w1c);
  k_rownorm<<<NTP, 256, 0, stream>>>(pn, tpp, U_N);
  k_rownorm<<<S_N, 256, 0, stream>>>(sn, u, U_N);
  k_cos<<<NTP * S_N, 128, 0, stream>>>(tpp, u, pn, sn, cosv);
  k_build_pu<<<GR((long)NTP * S_PAD * U_N), 256, 0, stream>>>(tpp, u, pu);
  k_inity<<<GR((long)NTP * S_N), 256, 0, stream>>>(y, mlpbo);
  k_gemm_y<<<dim3(48, 10, 5), 128, 0, stream>>>(y, pu, Wpr_bf, H1, basep, cosv, w1c, mlpWo);
}
